// PSF_85839216378384
// MI455X (gfx1250) — matrix operand order repaired and checked by reading, not compiled, not run on hardware
//
#include <hip/hip_runtime.h>

// ---------------------------------------------------------------------------
// PSF fused pipeline for MI455X (gfx1250, wave32, WMMA + async-LDS staging).
// B=8, C=256, H=W=128.  P = B*H*W = 131072 pixels.
//
// ws layout (bytes):
//   actA  (adain,  bf16, P*256)            @ 0          (67,108,864)
//   actF  (normF,  bf16, P*256)            @ 1*ACT
//   actB  (normB,  bf16, P*256)            @ 2*ACT
//   actH  (back,   bf16, P*256)            @ 3*ACT
//   EE    (f32,  P*256)                    @ 4*ACT      (134,217,728)
//   HH    (f32,  P*256)                    @ 4*ACT+EEB
//   S     (f32,  P)                        @ ...        (524,288)
//   minb/maxb (u32, 8+8)                   @ ...
//   wprep (bf16, 4*256*256, frag-major)    @ ...        (524,288)
// total ~537.9 MB
// ---------------------------------------------------------------------------

typedef __bf16 bf16_t;
typedef __attribute__((ext_vector_type(16))) __bf16 v16bf;
typedef __attribute__((ext_vector_type(8)))  __bf16 v8bf;
typedef __attribute__((ext_vector_type(8)))  float  v8f;

#define NB   8
#define NC   256
#define HW   16384      // 128*128
#define NP   131072     // NB*HW

// 32-bit LDS offset of a generic pointer into __shared__ (addrspace(3) cast).
#define LDS_OFF(p) \
  ((unsigned)(unsigned long long)(__attribute__((address_space(3))) const void*)(p))

__device__ __forceinline__ unsigned enc_ord(float f) {
  unsigned b = __float_as_uint(f);
  return (b & 0x80000000u) ? ~b : (b | 0x80000000u);
}
__device__ __forceinline__ float dec_ord(unsigned u) {
  unsigned b = (u & 0x80000000u) ? (u & 0x7FFFFFFFu) : ~u;
  return __uint_as_float(b);
}

// ---------------------------------------------------------------- init kernel
__global__ void psf_init_kernel(unsigned* __restrict__ minb,
                                unsigned* __restrict__ maxb) {
  int t = threadIdx.x;
  if (t < NB) { minb[t] = 0xFFFFFFFFu; maxb[t] = 0u; }
}

// --------------------------------------------------------------- stats kernel
// One block per (b, c, 32x32 tile). Separable 7x7 box sums with 3-halo in LDS.
// Writes pixel-major bf16 activations: act[(p)*256 + c].
__global__ __launch_bounds__(256) void psf_stats_kernel(
    const float* __restrict__ front, const float* __restrict__ back,
    bf16_t* __restrict__ actA, bf16_t* __restrict__ actF,
    bf16_t* __restrict__ actB, bf16_t* __restrict__ actH) {
  __shared__ float rawF[38][40], rawB[38][40];
  __shared__ float hsF[38][33], hs2F[38][33], hsB[38][33], hs2B[38][33];

  const int bid  = blockIdx.x;
  const int tile = bid & 15;
  const int c    = (bid >> 4) & 255;
  const int b    = bid >> 12;
  const int r0   = (tile >> 2) * 32;
  const int c0   = (tile & 3) * 32;
  const size_t chan = ((size_t)b * NC + c) * HW;

  // load 38x38 region (3-halo), zero padded
  for (int i = threadIdx.x; i < 38 * 38; i += 256) {
    int y = i / 38, x = i % 38;
    int gy = r0 + y - 3, gx = c0 + x - 3;
    bool ok = (gy >= 0) & (gy < 128) & (gx >= 0) & (gx < 128);
    size_t a = chan + (size_t)gy * 128 + gx;
    rawF[y][x] = ok ? front[a] : 0.f;
    rawB[y][x] = ok ? back[a]  : 0.f;
  }
  __syncthreads();

  // horizontal 7-tap sums (x and x^2) for 38 rows x 32 output cols
  for (int i = threadIdx.x; i < 38 * 32; i += 256) {
    int y = i >> 5, x = i & 31;
    float s = 0.f, s2 = 0.f, t = 0.f, t2 = 0.f;
#pragma unroll
    for (int d = 0; d < 7; ++d) {
      float f = rawF[y][x + d]; s += f; s2 += f * f;
      float g = rawB[y][x + d]; t += g; t2 += g * g;
    }
    hsF[y][x] = s; hs2F[y][x] = s2; hsB[y][x] = t; hs2B[y][x] = t2;
  }
  __syncthreads();

  // vertical 7-tap, stats, normalize, adain; store bf16 pixel-major
  for (int i = threadIdx.x; i < 1024; i += 256) {
    int y = i >> 5, x = i & 31;
    float sF = 0.f, s2F = 0.f, sB = 0.f, s2B = 0.f;
#pragma unroll
    for (int d = 0; d < 7; ++d) {
      sF += hsF[y + d][x];  s2F += hs2F[y + d][x];
      sB += hsB[y + d][x];  s2B += hs2B[y + d][x];
    }
    float meanF = sF * (1.f / 49.f);
    float varF  = fmaxf((s2F - 49.f * meanF * meanF) * (1.f / 48.f), 0.f);
    float stdF  = sqrtf(varF);
    float meanB = sB * (1.f / 49.f);
    float varB  = fmaxf((s2B - 49.f * meanB * meanB) * (1.f / 48.f), 0.f);
    float stdB  = sqrtf(varB);                 // no eps on y_std (as in source)
    float fval  = rawF[y + 3][x + 3];
    float bval  = rawB[y + 3][x + 3];
    float nF = (fval - meanF) / (stdF + 1e-8f);
    float nB = (bval - meanB) / (stdB + 1e-8f);
    float ad = nF * stdB + meanB;
    size_t p = (size_t)b * HW + (size_t)(r0 + y) * 128 + (c0 + x);
    size_t o = p * NC + c;
    actA[o] = (bf16_t)ad;
    actF[o] = (bf16_t)nF;
    actB[o] = (bf16_t)nB;
    actH[o] = (bf16_t)bval;
  }
}

// --------------------------------------------------------------- weight prep
// Pack four 256x256 f32 (O,C) weights into bf16 WMMA B-fragment-major layout:
// index = (((g*16 + nt)*8 + kc)*32 + lane)*16 + e ; n = nt*16 + lane%16 ;
// k = kc*32 + (lane/16)*8 + (e%8) + (e/8)*16  (K x N layout for v_wmma 16x16x32 bf16).
__global__ void psf_wprep_kernel(const float* __restrict__ we,
                                 const float* __restrict__ wf,
                                 const float* __restrict__ wg,
                                 const float* __restrict__ wh,
                                 bf16_t* __restrict__ wprep) {
  int t = blockIdx.x * 256 + threadIdx.x;     // 262144 total
  int e    = t & 15;
  int lane = (t >> 4) & 31;
  int kc   = (t >> 9) & 7;
  int g    = t >> 16;
  const float* w = (g == 0) ? we : (g == 1) ? wf : (g == 2) ? wg : wh;
  int n = (((t >> 12) & 15) << 4) + (lane & 15);
  int k = kc * 32 + (lane >> 4) * 8 + (e & 7) + (e >> 3) * 16;
  wprep[(size_t)t] = (bf16_t)w[n * NC + k];
}

// ----------------------------------------------------------- fused GEMM + S
// Block = 256 threads = 8 waves; covers 32 pixels x all 256 outputs x 4 GEMMs.
// Wave w owns ntiles {2w, 2w+1} and both 16-row M-subtiles, so each B fragment
// is reused for 2 WMMAs. 128 v_wmma_f32_16x16x32_bf16 per wave (K fully
// unrolled). Activation staging uses gfx1250 async LDS loads (ASYNCcnt):
// per (tensor, thread) the 4 staged 16B units step global and LDS by exactly
// 4 KB, so addresses fold into immediate offsets.
__global__ __launch_bounds__(256) void psf_gemm_kernel(
    const bf16_t* __restrict__ actA, const bf16_t* __restrict__ actF,
    const bf16_t* __restrict__ actB, const bf16_t* __restrict__ actH,
    const bf16_t* __restrict__ wprep,
    const float* __restrict__ biasE, const float* __restrict__ biasF,
    const float* __restrict__ biasG, const float* __restrict__ biasH,
    float* __restrict__ EE, float* __restrict__ HHo,
    float* __restrict__ S, unsigned* __restrict__ minb,
    unsigned* __restrict__ maxb) {
  __shared__ bf16_t lds[4][32][NC];           // 64 KB activation staging
  __shared__ float sdot[32], sff[32], sgg[32];

  const int tid   = threadIdx.x;
  const int pbase = blockIdx.x * 32;

  // async staging: 4 tensors x 32 pixels x 256 ch x 2B = 4096 x 16B units
  {
    const bf16_t* srcs[4] = {actA, actF, actB, actH};
    const int r0 = tid >> 5;                  // 0..7 (row group)
    const int cu = tid & 31;                  // 16B unit within row
#pragma unroll
    for (int g = 0; g < 4; ++g) {
      const bf16_t* base = srcs[g] + (size_t)(pbase + r0) * NC + cu * 8;
      unsigned lbase = LDS_OFF(&lds[g][r0][cu * 8]);
#pragma unroll
      for (int it = 0; it < 4; ++it) {        // rows r0, r0+8, r0+16, r0+24
        asm volatile("global_load_async_to_lds_b128 %0, %1, off"
                     :: "v"(lbase + (unsigned)(it * 4096)),
                        "v"(base + it * 2048) : "memory");
      }
    }
    asm volatile("s_wait_asynccnt 0x0" ::: "memory");
  }
  if (tid < 32) { sdot[tid] = 0.f; sff[tid] = 0.f; sgg[tid] = 0.f; }
  __syncthreads();

  const int wave = tid >> 5;
  const int lane = tid & 31;
  const int hl   = lane >> 4;                 // half-wave (K split of A frag)
  const int m    = lane & 15;                 // A row / C column index

  v8f acc[4][2][2];                           // [gemm][ntile][msub]
#pragma unroll
  for (int g = 0; g < 4; ++g)
#pragma unroll
    for (int t = 0; t < 2; ++t)
#pragma unroll
      for (int s = 0; s < 2; ++s)
        acc[g][t][s] = (v8f){0.f,0.f,0.f,0.f,0.f,0.f,0.f,0.f};

  const v16bf* wfr = (const v16bf*)wprep;

#pragma unroll
  for (int kc = 0; kc < 8; ++kc) {
    if (kc < 7)  // hint next K-chunk of B fragments into cache
      __builtin_prefetch(&wfr[(((wave * 2) * 8) + kc + 1) * 32 + lane], 0, 1);
#pragma unroll
    for (int g = 0; g < 4; ++g) {
      // A fragments for both 16-row M-subtiles
      v16bf a[2];
#pragma unroll
      for (int s = 0; s < 2; ++s) {
        v8bf lo = *(const v8bf*)&lds[g][s * 16 + m][kc * 32 + hl * 8];
        v8bf hi = *(const v8bf*)&lds[g][s * 16 + m][kc * 32 + 16 + hl * 8];
#pragma unroll
        for (int i = 0; i < 8; ++i) { a[s][i] = lo[i]; a[s][i + 8] = hi[i]; }
      }
#pragma unroll
      for (int t = 0; t < 2; ++t) {
        int nt = wave * 2 + t;
        v16bf bm = wfr[(((g * 16 + nt) * 8) + kc) * 32 + lane];  // reused 2x
#pragma unroll
        for (int s = 0; s < 2; ++s)
          acc[g][t][s] = __builtin_amdgcn_wmma_f32_16x16x32_bf16(
              false, a[s], false, bm, (short)0, acc[g][t][s], false, false);
      }
    }
  }

  // epilogue: bias, EE/HH stores, FF·GG reductions (ds_add_f32 atomics)
#pragma unroll
  for (int t = 0; t < 2; ++t) {
    int nt = wave * 2 + t;
    int n  = nt * 16 + m;
    float bE = biasE[n], bF = biasF[n], bG = biasG[n], bH = biasH[n];
#pragma unroll
    for (int s = 0; s < 2; ++s) {
#pragma unroll
      for (int e = 0; e < 8; ++e) {
        int row = s * 16 + 8 * hl + e;        // pixel within block (0..31)
        size_t q = (size_t)(pbase + row) * NC + n;
        float ee = acc[0][t][s][e] + bE;
        float ff = acc[1][t][s][e] + bF;
        float gg = acc[2][t][s][e] + bG;
        float hh = acc[3][t][s][e] + bH;
        EE[q]  = ee;
        HHo[q] = hh;
        atomicAdd(&sdot[row], ff * gg);
        atomicAdd(&sff[row],  ff * ff);
        atomicAdd(&sgg[row],  gg * gg);
      }
    }
  }
  __syncthreads();

  if (tid < 32) {
    float s = sdot[tid] / (sqrtf(sff[tid]) * sqrtf(sgg[tid]));
    int p = pbase + tid;
    S[p] = s;
    int b = p >> 14;                          // p / HW
    unsigned u = enc_ord(s);
    atomicMin(&minb[b], u);
    atomicMax(&maxb[b], u);
  }
}

// ----------------------------------------------------------------- fuse kernel
__global__ __launch_bounds__(256) void psf_fuse_kernel(
    const float* __restrict__ EE, const float* __restrict__ HHo,
    const float* __restrict__ S, const unsigned* __restrict__ minb,
    const unsigned* __restrict__ maxb, float* __restrict__ out) {
  int idx = blockIdx.x * 256 + threadIdx.x;   // over B*C*H*W = 33,554,432
  int hw = idx & (HW - 1);
  int c  = (idx >> 14) & 255;
  int b  = idx >> 22;
  int p  = b * HW + hw;
  float mn = dec_ord(minb[b]);
  float mx = dec_ord(maxb[b]);
  float sn = (S[p] - mn) / (mx - mn);
  size_t q = (size_t)p * NC + c;
  out[idx] = sn * EE[q] + (1.f - sn) * HHo[q];
}

// ---------------------------------------------------------------------------
extern "C" void kernel_launch(void* const* d_in, const int* in_sizes, int n_in,
                              void* d_out, int out_size, void* d_ws, size_t ws_size,
                              hipStream_t stream) {
  const float* front = (const float*)d_in[0];
  const float* back  = (const float*)d_in[1];
  // d_in[2] = mask (unused by the reference math)
  const float* we = (const float*)d_in[3];
  const float* be = (const float*)d_in[4];
  const float* wf = (const float*)d_in[5];
  const float* bf = (const float*)d_in[6];
  const float* wg = (const float*)d_in[7];
  const float* bg = (const float*)d_in[8];
  const float* wh = (const float*)d_in[9];
  const float* bh = (const float*)d_in[10];
  float* out = (float*)d_out;

  char* ws = (char*)d_ws;
  const size_t ACT = (size_t)NP * NC * 2;     // 67,108,864 B
  const size_t EEB = (size_t)NP * NC * 4;     // 134,217,728 B
  bf16_t* actA = (bf16_t*)(ws + 0 * ACT);
  bf16_t* actF = (bf16_t*)(ws + 1 * ACT);
  bf16_t* actB = (bf16_t*)(ws + 2 * ACT);
  bf16_t* actH = (bf16_t*)(ws + 3 * ACT);
  float*  EE   = (float*)(ws + 4 * ACT);
  float*  HHo  = (float*)(ws + 4 * ACT + EEB);
  float*  S    = (float*)(ws + 4 * ACT + 2 * EEB);
  unsigned* minb = (unsigned*)(ws + 4 * ACT + 2 * EEB + (size_t)NP * 4);
  unsigned* maxb = minb + NB;
  bf16_t* wprep = (bf16_t*)((char*)(minb + 16));

  psf_init_kernel<<<1, 32, 0, stream>>>(minb, maxb);
  psf_stats_kernel<<<NB * NC * 16, 256, 0, stream>>>(front, back,
                                                     actA, actF, actB, actH);
  psf_wprep_kernel<<<1024, 256, 0, stream>>>(we, wf, wg, wh, wprep);
  psf_gemm_kernel<<<NP / 32, 256, 0, stream>>>(actA, actF, actB, actH, wprep,
                                               be, bf, bg, bh,
                                               EE, HHo, S, minb, maxb);
  psf_fuse_kernel<<<(NB * NC * HW) / 256, 256, 0, stream>>>(EE, HHo, S,
                                                            minb, maxb, out);
}
